// PointNet_ssg_Plus_Plus_14336600834839
// MI455X (gfx1250) — compile-verified, matrix operand order checked by reading, not run on hardware
//
#include <hip/hip_runtime.h>
#include <hip/hip_bf16.h>
#include <cstdint>

// ---------------------------------------------------------------------------
// Types for CDNA5 WMMA (wave32).  D = A(16x32 f16) * B(32x16 f16) + C(f32)
// ---------------------------------------------------------------------------
typedef __attribute__((ext_vector_type(16))) _Float16 v16h;
typedef __attribute__((ext_vector_type(8)))  _Float16 v8h;
typedef __attribute__((ext_vector_type(8)))  float    v8f;

// Per-lane fragment load for the 16-bit 16x32 A layout (and its B mirror):
// lane<16 needs halves [k..k+7] and [k+16..k+23]; lane>=16 uses base+8.
// Both 8-half segments are contiguous 16B -> two b128 loads.
__device__ __forceinline__ v16h load_frag(const _Float16* p) {
    v8h lo = *(const v8h*)(p);
    v8h hi = *(const v8h*)(p + 16);
    return __builtin_shufflevector(lo, hi, 0,1,2,3,4,5,6,7,8,9,10,11,12,13,14,15);
}

__device__ __forceinline__ v8f wmma_f16(v16h a, v16h b, v8f c) {
    return __builtin_amdgcn_wmma_f32_16x16x32_f16(
        /*neg_a=*/false, a, /*neg_b=*/false, b,
        /*c_mod=*/(short)0, c, /*reuse_a=*/false, /*reuse_b=*/false);
}

// ---------------------------------------------------------------------------
// Weight conversion: f32 (Cout, Cin) -> f16 (Cout, Kpad), zero-padded K tail.
// ---------------------------------------------------------------------------
__global__ void convert_w_kernel(const float* __restrict__ W, _Float16* __restrict__ out,
                                 int cout, int cin, int kpad) {
    int i = blockIdx.x * blockDim.x + threadIdx.x;
    int total = cout * kpad;
    if (i >= total) return;
    int c = i / kpad, k = i % kpad;
    out[i] = (k < cin) ? (_Float16)W[c * cin + k] : (_Float16)0.0f;
}

// ---------------------------------------------------------------------------
// Farthest point sampling: one block per batch, dists in registers.
// Matches reference: emits `last` each step, starting at 0; first-max on ties.
// ---------------------------------------------------------------------------
#define FPS_T 256
__global__ void fps_kernel(const float* __restrict__ xyz, int n, int npoint,
                           int* __restrict__ out) {
    int b = blockIdx.x, tid = threadIdx.x;
    const float* p = xyz + (size_t)b * n * 3;
    int* o = out + (size_t)b * npoint;
    int ppt = (n + FPS_T - 1) / FPS_T;           // <= 16 for n<=4096
    float dist[16];
    #pragma unroll
    for (int i = 0; i < 16; i++) dist[i] = 1.0e10f;
    __shared__ float sval[FPS_T];
    __shared__ int   sidx[FPS_T];
    int last = 0;
    for (int it = 0; it < npoint; it++) {
        if (tid == 0) o[it] = last;
        float cx = p[last * 3 + 0], cy = p[last * 3 + 1], cz = p[last * 3 + 2];
        float best = -1.0f; int bi = n;
        for (int i = 0; i < ppt; i++) {
            int pt = tid + i * FPS_T;
            if (pt < n) {
                float dx = p[pt * 3 + 0] - cx, dy = p[pt * 3 + 1] - cy, dz = p[pt * 3 + 2] - cz;
                float d = dx * dx + dy * dy + dz * dz;
                float dm = fminf(dist[i], d);
                dist[i] = dm;
                if (dm > best || (dm == best && pt < bi)) { best = dm; bi = pt; }
            }
        }
        sval[tid] = best; sidx[tid] = bi;
        __syncthreads();
        for (int s = FPS_T / 2; s > 0; s >>= 1) {
            if (tid < s) {
                if (sval[tid + s] > sval[tid] ||
                    (sval[tid + s] == sval[tid] && sidx[tid + s] < sidx[tid])) {
                    sval[tid] = sval[tid + s]; sidx[tid] = sidx[tid + s];
                }
            }
            __syncthreads();
        }
        last = sidx[0];
        __syncthreads();
    }
}

// Gather 3-vector centers: out[b,m,:] = xyz[b, idx[b,m], :]
__global__ void gather3_kernel(const float* __restrict__ xyz, const int* __restrict__ idx,
                               float* __restrict__ out, int n, int npoint, int total) {
    int i = blockIdx.x * blockDim.x + threadIdx.x;
    if (i >= total) return;
    int b = i / npoint;
    int id = idx[i];
    const float* s = xyz + ((size_t)b * n + id) * 3;
    out[(size_t)i * 3 + 0] = s[0];
    out[(size_t)i * 3 + 1] = s[1];
    out[(size_t)i * 3 + 2] = s[2];
}

// ---------------------------------------------------------------------------
// Ball query: one wave32 per center; ordered append via ballot+popc prefix.
// Reproduces "first nsample in index order, pad with first hit, else 0".
// ---------------------------------------------------------------------------
__global__ void ball_query_kernel(const float* __restrict__ xyz, const float* __restrict__ ctr,
                                  int n, int npoint, float r2,
                                  int* __restrict__ nidx, int total_centers) {
    int gw = (blockIdx.x * blockDim.x + threadIdx.x) >> 5;
    int lane = threadIdx.x & 31;
    if (gw >= total_centers) return;
    int b = gw / npoint;
    const float* p = xyz + (size_t)b * n * 3;
    float cx = ctr[(size_t)gw * 3 + 0], cy = ctr[(size_t)gw * 3 + 1], cz = ctr[(size_t)gw * 3 + 2];
    int* out = nidx + (size_t)gw * 64;
    int count = 0, first = 0;
    for (int base = 0; base < n && count < 64; base += 32) {
        int pt = base + lane;
        bool in = false;
        if (pt < n) {
            float dx = p[pt * 3 + 0] - cx, dy = p[pt * 3 + 1] - cy, dz = p[pt * 3 + 2] - cz;
            in = (dx * dx + dy * dy + dz * dz) <= r2;
        }
        unsigned mask = (unsigned)__ballot(in);            // wave32: low 32 bits
        if (in) {
            int pos = count + __popc(mask & ((1u << lane) - 1u));
            if (pos < 64) out[pos] = pt;
        }
        if (count == 0 && mask) first = base + __ffs(mask) - 1;
        count += __popc(mask);
    }
    if (count < 64) {
        int fill = (count > 0) ? first : 0;
        for (int j = count + lane; j < 64; j += 32) out[j] = fill;
    }
}

// ---------------------------------------------------------------------------
// One MLP layer on an LDS-resident 64-row tile.  8 waves split the
// (4 M-tiles) x (Cout/16 N-tiles) work; A frags from LDS, B frags from L2.
// ---------------------------------------------------------------------------
__device__ __forceinline__ void mlp_layer(const _Float16* bin, _Float16* bout,
                                          const _Float16* __restrict__ W,
                                          const float* __restrict__ g,
                                          const float* __restrict__ bb,
                                          int K, int Cout, int lane, int wave, int ldsw) {
    int ntiles = Cout >> 4;
    int ak = (lane < 16) ? 0 : 8;
    for (int t = wave; t < 4 * ntiles; t += 8) {
        int mt = t & 3, nt = t >> 2;
        v8f acc = {};
        int arow = mt * 16 + (lane & 15);
        int bcol = nt * 16 + (lane & 15);
        const _Float16* ap = bin + arow * ldsw + ak;
        const _Float16* bp = W + (size_t)bcol * K + ak;
        for (int kk = 0; kk < K; kk += 32) {
            v16h a  = load_frag(ap + kk);
            v16h bf = load_frag(bp + kk);
            acc = wmma_f16(a, bf, acc);
        }
        float gv = g[bcol], bv = bb[bcol];
        int rbase = mt * 16 + ((lane < 16) ? 0 : 8);
        #pragma unroll
        for (int r = 0; r < 8; r++) {
            float y = fmaxf(gv * acc[r] + bv, 0.0f);
            bout[(rbase + r) * ldsw + bcol] = (_Float16)y;
        }
    }
}

// ---------------------------------------------------------------------------
// Fused SA stage: one block per (batch, center).  Builds grouped 64 x Cin tile
// in LDS (rel-xyz ++ feats), runs 3 WMMA MLP layers ping-ponging LDS, then
// max-pools over the 64 samples.  No global intermediates.
// ---------------------------------------------------------------------------
__global__ void fused_sa_kernel(const float* __restrict__ xyz, const _Float16* __restrict__ feats,
                                const int* __restrict__ nidx, const float* __restrict__ new_xyz,
                                int n, int npoint, int cf, int ldsw,
                                const _Float16* W0, const float* g0, const float* b0,
                                int k0, int c0,
                                const _Float16* W1, const float* g1, const float* b1,
                                const _Float16* W2, const float* g2, const float* b2,
                                int c1, int c2,
                                _Float16* __restrict__ pooled) {
    extern __shared__ __align__(16) _Float16 smem[];
    _Float16* buf0 = smem;
    _Float16* buf1 = smem + 64 * ldsw;
    int tid = threadIdx.x, lane = tid & 31, wave = tid >> 5;
    int center = blockIdx.x;
    int b = center / npoint;
    int cin = 3 + cf;

    // Zero the K pad of the input tile (WMMA NaN-propagates, pads must be 0).
    int padw = k0 - cin;
    if (padw > 0) {
        for (int i = tid; i < 64 * padw; i += 256) {
            int s = i / padw, c = cin + i % padw;
            buf0[s * ldsw + c] = (_Float16)0.0f;
        }
    }
    // Build grouped rows: rel xyz then gathered features.
    const float* cp = new_xyz + (size_t)center * 3;
    float ccx = cp[0], ccy = cp[1], ccz = cp[2];
    const int* ni = nidx + (size_t)center * 64;
    for (int i = tid; i < 64 * cin; i += 256) {
        int s = i / cin, c = i % cin;
        int id = ni[s];
        float v;
        if (c < 3) {
            const float* pp = xyz + ((size_t)b * n + id) * 3;
            v = pp[c] - ((c == 0) ? ccx : (c == 1) ? ccy : ccz);
        } else {
            v = (float)feats[((size_t)b * n + id) * cf + (c - 3)];
        }
        buf0[s * ldsw + c] = (_Float16)v;
    }
    __syncthreads();
    mlp_layer(buf0, buf1, W0, g0, b0, k0, c0, lane, wave, ldsw);
    __syncthreads();
    mlp_layer(buf1, buf0, W1, g1, b1, c0, c1, lane, wave, ldsw);
    __syncthreads();
    mlp_layer(buf0, buf1, W2, g2, b2, c1, c2, lane, wave, ldsw);
    __syncthreads();
    // Max-pool over the 64 samples.
    for (int c = tid; c < c2; c += 256) {
        float mx = -3.0e38f;
        for (int s = 0; s < 64; s++) mx = fmaxf(mx, (float)buf1[s * ldsw + c]);
        pooled[(size_t)center * c2 + c] = (_Float16)mx;
    }
}

// ---------------------------------------------------------------------------
// Generic WMMA GEMM with fused epilogue:  Y = act(X * W^T)
//   mode 0: f16 out, y = relu(g*dot + b)         (BN-eval + ReLU)
//   mode 1: f32 out, y = dot + b                 (final classifier)
// Block = 4 waves; each wave owns one 16x16 tile of a 16 x 64 block tile.
// ---------------------------------------------------------------------------
__global__ void gemm_wmma_kernel(const _Float16* __restrict__ X, int ldx,
                                 const _Float16* __restrict__ W, int K,
                                 const float* __restrict__ g, const float* __restrict__ bias,
                                 int Cout, int ldy, int mode,
                                 _Float16* __restrict__ Yh, float* __restrict__ Yf) {
    int lane = threadIdx.x & 31, wave = threadIdx.x >> 5;
    int mbase = blockIdx.x * 16;
    int nbase = blockIdx.y * 64 + wave * 16;
    int ak = (lane < 16) ? 0 : 8;
    int arow = mbase + (lane & 15);
    int bcol = nbase + (lane & 15);
    int bcolc = (bcol < Cout) ? bcol : 0;          // clamp; junk cols discarded
    v8f acc = {};
    const _Float16* ap = X + (size_t)arow * ldx + ak;
    const _Float16* bp = W + (size_t)bcolc * K + ak;
    for (int kk = 0; kk < K; kk += 32) {
        v16h a  = load_frag(ap + kk);
        v16h bf = load_frag(bp + kk);
        acc = wmma_f16(a, bf, acc);
    }
    int col = nbase + (lane & 15);
    int rbase = mbase + ((lane < 16) ? 0 : 8);
    if (mode == 0) {
        float gv = 0.0f, bv = 0.0f;
        if (col < Cout) { gv = g[col]; bv = bias[col]; }
        #pragma unroll
        for (int r = 0; r < 8; r++) {
            if (col < Cout) {
                float y = fmaxf(gv * acc[r] + bv, 0.0f);
                Yh[(size_t)(rbase + r) * ldy + col] = (_Float16)y;
            } else if (col < ldy) {
                Yh[(size_t)(rbase + r) * ldy + col] = (_Float16)0.0f;
            }
        }
    } else {
        if (col < Cout) {
            float bv = bias[col];
            #pragma unroll
            for (int r = 0; r < 8; r++)
                Yf[(size_t)(rbase + r) * Cout + col] = acc[r] + bv;
        }
    }
}

// Group-all concat for SA3: X[row] = [xyz(3) | feats(cf) | zero pad]
__global__ void group_all_kernel(const float* __restrict__ xyz, const _Float16* __restrict__ feats,
                                 int cf, int kpad, _Float16* __restrict__ X, int total) {
    int i = blockIdx.x * blockDim.x + threadIdx.x;
    if (i >= total) return;
    int row = i / kpad, c = i % kpad;
    float v;
    if (c < 3)             v = xyz[(size_t)row * 3 + c];
    else if (c < 3 + cf)   v = (float)feats[(size_t)row * cf + (c - 3)];
    else                   v = 0.0f;
    X[i] = (_Float16)v;
}

// Max over `rows` consecutive rows per group (SA3 global pool).
__global__ void maxpool_rows_kernel(const _Float16* __restrict__ Y, int rows, int C, int ldy,
                                    _Float16* __restrict__ out, int total) {
    int i = blockIdx.x * blockDim.x + threadIdx.x;
    if (i >= total) return;
    int gidx = i / C, c = i % C;
    const _Float16* p = Y + (size_t)gidx * rows * ldy + c;
    float mx = -3.0e38f;
    for (int r = 0; r < rows; r++) mx = fmaxf(mx, (float)p[(size_t)r * ldy]);
    out[i] = (_Float16)mx;
}

// ---------------------------------------------------------------------------
// Host-side launch
// ---------------------------------------------------------------------------
static const int BB = 16, NN = 4096, CLSN = 40;

extern "C" void kernel_launch(void* const* d_in, const int* in_sizes, int n_in,
                              void* d_out, int out_size, void* d_ws, size_t ws_size,
                              hipStream_t stream) {
    (void)in_sizes; (void)n_in; (void)out_size; (void)ws_size;
    const float* pc = (const float*)d_in[0];
    // param flattening: sa1 W,g,b x3 @1..9; sa2 @10..18; sa3 @19..27;
    // fc (W,g,b),(W,g,b),(W,b) @28..35
    auto F = [&](int i) { return (const float*)d_in[i]; };

    // workspace allocator
    size_t off = 0;
    auto alloc = [&](size_t bytes) {
        off = (off + 255) & ~(size_t)255;
        void* p = (char*)d_ws + off;
        off += bytes;
        return p;
    };
    auto allocH = [&](size_t elems) { return (_Float16*)alloc(elems * sizeof(_Float16)); };
    auto allocF = [&](size_t elems) { return (float*)alloc(elems * sizeof(float)); };
    auto allocI = [&](size_t elems) { return (int*)alloc(elems * sizeof(int)); };

    // f16 zero-K-padded weights
    _Float16* w10 = allocH((size_t)64 * 32);
    _Float16* w11 = allocH((size_t)64 * 64);
    _Float16* w12 = allocH((size_t)128 * 64);
    _Float16* w20 = allocH((size_t)128 * 160);
    _Float16* w21 = allocH((size_t)128 * 128);
    _Float16* w22 = allocH((size_t)256 * 128);
    _Float16* w30 = allocH((size_t)256 * 288);
    _Float16* w31 = allocH((size_t)512 * 256);
    _Float16* w32 = allocH((size_t)1024 * 512);
    _Float16* wf1 = allocH((size_t)512 * 1024);
    _Float16* wf2 = allocH((size_t)256 * 512);
    _Float16* wf3 = allocH((size_t)CLSN * 256);

    int*      cidx1 = allocI((size_t)BB * 512);
    float*    nxyz1 = allocF((size_t)BB * 512 * 3);
    int*      nidx1 = allocI((size_t)BB * 512 * 64);
    _Float16* pool1 = allocH((size_t)BB * 512 * 128);
    int*      cidx2 = allocI((size_t)BB * 128);
    float*    nxyz2 = allocF((size_t)BB * 128 * 3);
    int*      nidx2 = allocI((size_t)BB * 128 * 64);
    _Float16* pool2 = allocH((size_t)BB * 128 * 256);
    _Float16* X3    = allocH((size_t)BB * 128 * 288);
    _Float16* Y3a   = allocH((size_t)BB * 128 * 256);
    _Float16* Y3b   = allocH((size_t)BB * 128 * 512);
    _Float16* Y3c   = allocH((size_t)BB * 128 * 1024);
    _Float16* pool3 = allocH((size_t)BB * 1024);
    _Float16* fy1   = allocH((size_t)BB * 512);
    _Float16* fy2   = allocH((size_t)BB * 256);

    auto cvt = [&](const float* W, _Float16* o, int cout, int cin, int kpad) {
        int tot = cout * kpad;
        convert_w_kernel<<<(tot + 255) / 256, 256, 0, stream>>>(W, o, cout, cin, kpad);
    };
    cvt(F(1),  w10, 64,   3,   32);  cvt(F(4),  w11, 64,   64,  64);
    cvt(F(7),  w12, 128,  64,  64);
    cvt(F(10), w20, 128,  131, 160); cvt(F(13), w21, 128,  128, 128);
    cvt(F(16), w22, 256,  128, 128);
    cvt(F(19), w30, 256,  259, 288); cvt(F(22), w31, 512,  256, 256);
    cvt(F(25), w32, 1024, 512, 512);
    cvt(F(28), wf1, 512,  1024, 1024); cvt(F(31), wf2, 256, 512, 512);
    cvt(F(34), wf3, CLSN, 256, 256);

    // ---- SA1: 4096 -> 512 centers, r=0.2, ns=64, MLP 3->64->64->128 ----
    fps_kernel<<<BB, FPS_T, 0, stream>>>(pc, NN, 512, cidx1);
    {
        int tot = BB * 512;
        gather3_kernel<<<(tot + 255) / 256, 256, 0, stream>>>(pc, cidx1, nxyz1, NN, 512, tot);
        ball_query_kernel<<<(tot * 32 + 255) / 256, 256, 0, stream>>>(
            pc, nxyz1, NN, 512, 0.04f, nidx1, tot);
        size_t lds = (size_t)2 * 64 * 128 * sizeof(_Float16);   // 32 KB
        fused_sa_kernel<<<tot, 256, lds, stream>>>(
            pc, (const _Float16*)nullptr, nidx1, nxyz1, NN, 512, /*cf=*/0, /*ldsw=*/128,
            w10, F(2), F(3), /*k0=*/32, /*c0=*/64,
            w11, F(5), F(6),
            w12, F(8), F(9), /*c1=*/64, /*c2=*/128,
            pool1);
    }
    // ---- SA2: 512 -> 128 centers, r=0.4, ns=64, MLP 131->128->128->256 ----
    fps_kernel<<<BB, FPS_T, 0, stream>>>(nxyz1, 512, 128, cidx2);
    {
        int tot = BB * 128;
        gather3_kernel<<<(tot + 255) / 256, 256, 0, stream>>>(nxyz1, cidx2, nxyz2, 512, 128, tot);
        ball_query_kernel<<<(tot * 32 + 255) / 256, 256, 0, stream>>>(
            nxyz1, nxyz2, 512, 128, 0.16f, nidx2, tot);
        size_t lds = (size_t)2 * 64 * 256 * sizeof(_Float16);   // 64 KB
        fused_sa_kernel<<<tot, 256, lds, stream>>>(
            nxyz1, pool1, nidx2, nxyz2, 512, 128, /*cf=*/128, /*ldsw=*/256,
            w20, F(11), F(12), /*k0=*/160, /*c0=*/128,
            w21, F(14), F(15),
            w22, F(17), F(18), /*c1=*/128, /*c2=*/256,
            pool2);
    }
    // ---- SA3 (group all): concat xyz+feats (259->288), MLP ->256->512->1024, pool over 128 ----
    {
        int rows = BB * 128;
        int tot = rows * 288;
        group_all_kernel<<<(tot + 255) / 256, 256, 0, stream>>>(nxyz2, pool2, 256, 288, X3, tot);
        gemm_wmma_kernel<<<dim3(rows / 16, 256 / 64), 128, 0, stream>>>(
            X3, 288, w30, 288, F(20), F(21), 256, 256, 0, Y3a, nullptr);
        gemm_wmma_kernel<<<dim3(rows / 16, 512 / 64), 128, 0, stream>>>(
            Y3a, 256, w31, 256, F(23), F(24), 512, 512, 0, Y3b, nullptr);
        gemm_wmma_kernel<<<dim3(rows / 16, 1024 / 64), 128, 0, stream>>>(
            Y3b, 512, w32, 512, F(26), F(27), 1024, 1024, 0, Y3c, nullptr);
        int ptot = BB * 1024;
        maxpool_rows_kernel<<<(ptot + 255) / 256, 256, 0, stream>>>(
            Y3c, 128, 1024, 1024, pool3, ptot);
    }
    // ---- FC head: 1024->512->256->40 ----
    gemm_wmma_kernel<<<dim3(1, 512 / 64), 128, 0, stream>>>(
        pool3, 1024, wf1, 1024, F(29), F(30), 512, 512, 0, fy1, nullptr);
    gemm_wmma_kernel<<<dim3(1, 256 / 64), 128, 0, stream>>>(
        fy1, 512, wf2, 512, F(32), F(33), 256, 256, 0, fy2, nullptr);
    gemm_wmma_kernel<<<dim3(1, 1), 128, 0, stream>>>(
        fy2, 256, wf3, 256, nullptr, F(35), CLSN, CLSN, 1, nullptr, (float*)d_out);
}
